// Basic_RT_57750130262767
// MI455X (gfx1250) — matrix operand order repaired and checked by reading, not compiled, not run on hardware
//
#include <hip/hip_runtime.h>
#include <hip/hip_bf16.h>
#include <stddef.h>
#include <stdint.h>

// ---------------------------------------------------------------------------
// Feature probe: gfx1250 async global->LDS copy (ASYNCcnt path)
// Probe round confirmed the builtin exists and takes pointers to int4 vectors.
// ---------------------------------------------------------------------------
#ifndef __has_builtin
#define __has_builtin(x) 0
#endif
#if __has_builtin(__builtin_amdgcn_global_load_async_to_lds_b128) && \
    __has_builtin(__builtin_amdgcn_s_wait_asynccnt)
#define USE_ASYNC_LDS 1
typedef __attribute__((ext_vector_type(4))) int v4i_t;
typedef __attribute__((address_space(1))) v4i_t gv4i_t;   // global int4
typedef __attribute__((address_space(3))) v4i_t lv4i_t;   // LDS int4
#else
#define USE_ASYNC_LDS 0
#endif

// ---------------------------------------------------------------------------
// Types / WMMA helpers (CDNA5 gfx1250, wave32)
// ---------------------------------------------------------------------------
typedef __bf16 bf16_t;
typedef __attribute__((ext_vector_type(16))) __bf16 v16bf;
typedef __attribute__((ext_vector_type(8)))  float  v8f;

__device__ __forceinline__ v8f vzero8() {
    v8f z;
#pragma unroll
    for (int i = 0; i < 8; ++i) z[i] = 0.0f;
    return z;
}

// A fragment (16x32 bf16) from row-major bf16 in LDS, leading dim ldk.
__device__ __forceinline__ v16bf load_a_frag(const bf16_t* A, int ldk, int row,
                                             int khalf, int kbase) {
    const bf16_t* p0 = A + (size_t)row * ldk + kbase + khalf * 8;
    const bf16_t* p1 = p0 + 16;
    v16bf f;
#pragma unroll
    for (int t = 0; t < 8; ++t) f[t] = p0[t];
#pragma unroll
    for (int t = 0; t < 8; ++t) f[8 + t] = p1[t];
    return f;
}

// B fragment (32x16 bf16) from TRANSPOSED weights Wt[n][k], leading dim ldk.
__device__ __forceinline__ v16bf load_b_frag(const bf16_t* Wt, int ldk, int col,
                                             int khalf, int kbase) {
    const bf16_t* p = Wt + (size_t)col * ldk + kbase + khalf * 8;
    v16bf f;
#pragma unroll
    for (int t = 0; t < 8; ++t) f[t] = p[t];
#pragma unroll
    for (int t = 0; t < 8; ++t) f[8 + t] = p[16 + t];
    return f;
}

__device__ __forceinline__ v8f wmma_bf16(v16bf a, v16bf b, v8f c) {
    return __builtin_amdgcn_wmma_f32_16x16x32_bf16(false, a, false, b,
                                                   (short)0, c, false, false);
}

// ---------------------------------------------------------------------------
// K0: graph projections gQ/gK/gV (M=2 -> VALU). grid(2), block(128)
// ---------------------------------------------------------------------------
__global__ void graph_proj_kernel(const float* __restrict__ graph,
                                  const float* Wq, const float* bq,
                                  const float* Wk, const float* bk,
                                  const float* Wv, const float* bv,
                                  float* gQp, float* gKp, float* gVp) {
    int b = blockIdx.x, col = threadIdx.x;
    float q = bq[col], k = bk[col], v = bv[col];
    for (int kk = 0; kk < 128; ++kk) {
        float x = graph[b * 128 + kk];
        q += x * Wq[kk * 128 + col];
        k += x * Wk[kk * 128 + col];
        v += x * Wv[kk * 128 + col];
    }
    gQp[b * 128 + col] = q;
    gKp[b * 128 + col] = k;
    gVp[b * 128 + col] = v;
}

// ---------------------------------------------------------------------------
// Generic bf16-WMMA GEMM: C[M,N] = act(A[M,K] @ W[K,N] + bias)
// grid(M/128, N/128), block(256), dyn LDS 64KB
// ---------------------------------------------------------------------------
__global__ void gemm_wmma_kernel(const float* __restrict__ A,
                                 const float* __restrict__ W,
                                 const float* __restrict__ bias,
                                 float* __restrict__ C,
                                 int M, int N, int K, int relu) {
    extern __shared__ char smem[];
    bf16_t* sA  = (bf16_t*)smem;
    bf16_t* sWt = (bf16_t*)(smem + 32768);
    int tid = threadIdx.x, wave = tid >> 5, lane = tid & 31;
    int row0 = blockIdx.x * 128, col0 = blockIdx.y * 128;
    int colb = lane & 15, khalf = lane >> 4;

    v8f acc[8];
#pragma unroll
    for (int nt = 0; nt < 8; ++nt) acc[nt] = vzero8();

    for (int k0 = 0; k0 < K; k0 += 128) {
        for (int idx = tid; idx < 128 * 128; idx += 256) {
            int r = idx >> 7, kk = idx & 127;
            sA[idx] = (bf16_t)A[(size_t)(row0 + r) * K + k0 + kk];
        }
        for (int idx = tid; idx < 128 * 128; idx += 256) {
            int c = idx >> 7, kk = idx & 127;
            sWt[idx] = (bf16_t)W[(size_t)(k0 + kk) * N + col0 + c];
        }
        __syncthreads();
#pragma unroll
        for (int ks = 0; ks < 4; ++ks) {
            v16bf af = load_a_frag(sA, 128, wave * 16 + (lane & 15), khalf, ks * 32);
#pragma unroll
            for (int nt = 0; nt < 8; ++nt) {
                v16bf bfg = load_b_frag(sWt, 128, nt * 16 + colb, khalf, ks * 32);
                acc[nt] = wmma_bf16(af, bfg, acc[nt]);
            }
        }
        __syncthreads();
    }
#pragma unroll
    for (int nt = 0; nt < 8; ++nt) {
        int col = col0 + nt * 16 + colb;
        float bb = bias ? bias[col] : 0.0f;
#pragma unroll
        for (int g = 0; g < 8; ++g) {
            int row = row0 + wave * 16 + khalf * 8 + g;
            float v = acc[nt][g] + bb;
            if (relu) v = v > 0.0f ? v : 0.0f;
            C[(size_t)row * N + col] = v;
        }
    }
}

// ---------------------------------------------------------------------------
// Qf[b,i,j,:] = Weq*edge[b,i,j] + bias + nQp[b,i,:] + gQp[b,:]
// grid(256), block(256), dyn LDS 66048
// ---------------------------------------------------------------------------
__global__ void gemm_q_kernel(const float* __restrict__ edge,
                              const float* __restrict__ W,
                              const float* __restrict__ bias,
                              const float* __restrict__ nQp,
                              const float* __restrict__ gQp,
                              float* __restrict__ Qf) {
    extern __shared__ char smem[];
    bf16_t* sA   = (bf16_t*)smem;
    bf16_t* sWt  = (bf16_t*)(smem + 32768);
    float*  sAdd = (float*)(smem + 65536);
    int tid = threadIdx.x, wave = tid >> 5, lane = tid & 31;
    int bi = blockIdx.x, b = bi >> 7;
    int colb = lane & 15, khalf = lane >> 4;
    const float* Ar = edge + (size_t)bi * 16384;

    for (int idx = tid; idx < 16384; idx += 256) sA[idx] = (bf16_t)Ar[idx];
    for (int idx = tid; idx < 16384; idx += 256) {
        int c = idx >> 7, kk = idx & 127;
        sWt[idx] = (bf16_t)W[kk * 128 + c];
    }
    for (int c = tid; c < 128; c += 256)
        sAdd[c] = bias[c] + nQp[(size_t)bi * 128 + c] + gQp[b * 128 + c];
    __syncthreads();

    v8f acc[8];
#pragma unroll
    for (int nt = 0; nt < 8; ++nt) acc[nt] = vzero8();
#pragma unroll
    for (int ks = 0; ks < 4; ++ks) {
        v16bf af = load_a_frag(sA, 128, wave * 16 + (lane & 15), khalf, ks * 32);
#pragma unroll
        for (int nt = 0; nt < 8; ++nt) {
            v16bf bfg = load_b_frag(sWt, 128, nt * 16 + colb, khalf, ks * 32);
            acc[nt] = wmma_bf16(af, bfg, acc[nt]);
        }
    }
#pragma unroll
    for (int nt = 0; nt < 8; ++nt) {
        int col = nt * 16 + colb;
#pragma unroll
        for (int g = 0; g < 8; ++g) {
            int row = wave * 16 + khalf * 8 + g;
            Qf[(size_t)bi * 16384 + row * 128 + col] = acc[nt][g] + sAdd[col];
        }
    }
}

// ---------------------------------------------------------------------------
// Build augmented transposed bf16 weights WkT/WvT (128 cols x 256 k:
// k<128 -> Wek/Wev, k>=128 -> Wnk/Wnv) and column-add vectors.
// grid(32), block(256)
// ---------------------------------------------------------------------------
__global__ void prep_attn_kernel(const float* Wek_w, const float* Wnk_w,
                                 const float* Wev_w, const float* Wnv_w,
                                 const float* Wek_b, const float* Wnk_b,
                                 const float* Wev_b, const float* Wnv_b,
                                 const float* gKp, const float* gVp,
                                 bf16_t* WkT, bf16_t* WvT,
                                 float* kcolv, float* foldV) {
    int idx0 = blockIdx.x * blockDim.x + threadIdx.x;
    for (int idx = idx0; idx < 32768; idx += gridDim.x * blockDim.x) {
        int col = idx >> 8, k = idx & 255;
        float wk = (k < 128) ? Wek_w[k * 128 + col] : Wnk_w[(k - 128) * 128 + col];
        float wv = (k < 128) ? Wev_w[k * 128 + col] : Wnv_w[(k - 128) * 128 + col];
        WkT[idx] = (bf16_t)wk;
        WvT[idx] = (bf16_t)wv;
    }
    if (idx0 < 256) {
        int col = idx0 & 127;
        kcolv[idx0] = gKp[idx0] + Wnk_b[col] + Wek_b[col];
        foldV[idx0] = gVp[idx0] + Wnv_b[col] + Wev_b[col];
    }
}

// ---------------------------------------------------------------------------
// q1[b,i,j,h] = sum_d Qf[b,i,j,h*16+d] * kcolv[b,h*16+d]  (rank-1 K add folded)
// grid(256), block(256)
// ---------------------------------------------------------------------------
__global__ void q1_kernel(const float* __restrict__ Qf,
                          const float* __restrict__ kcolv,
                          float* __restrict__ q1) {
    __shared__ float sk[128];
    int bi = blockIdx.x, tid = threadIdx.x, b = bi >> 7;
    if (tid < 128) sk[tid] = kcolv[b * 128 + tid];
    __syncthreads();
    for (int idx = tid; idx < 1024; idx += 256) {
        int j = idx >> 3, h = idx & 7;
        const float* q = Qf + (size_t)bi * 16384 + (size_t)j * 128 + h * 16;
        float s = 0.0f;
#pragma unroll
        for (int d = 0; d < 16; ++d) s += q[d] * sk[h * 16 + d];
        q1[((size_t)bi * 128 + j) * 8 + h] = s;
    }
}

// ---------------------------------------------------------------------------
// Fused flash-attention over (j,c): one WG per (b,i), wave w owns bank c=w.
// bank1 streamed from HBM once; K/V are K=256 augmented WMMA GEMMs; online
// softmax in registers.  Weight + Q-tile staging uses async global->LDS copies
// (ASYNCcnt) when the toolchain exposes them.  dyn LDS 144384
// ---------------------------------------------------------------------------
__global__ void fused_attn_kernel(const float* __restrict__ bank1,
                                  const float* __restrict__ bank0,
                                  const bf16_t* __restrict__ WkT,   // WvT follows contiguously
                                  const bf16_t* __restrict__ WvT,
                                  const float* __restrict__ Qf,
                                  const float* __restrict__ q1,
                                  const float* __restrict__ foldV,
                                  float* __restrict__ attw) {
    extern __shared__ char smem[];
    bf16_t* sWk  = (bf16_t*)smem;              // 128x256 bf16 (64KB)
    bf16_t* sWv  = (bf16_t*)(smem + 65536);    // 128x256 bf16 (64KB)
    float*  sQ   = (float*)(smem + 131072);    // 16x128 (8KB)
    float*  sQ1  = (float*)(smem + 139264);    // 16x8
    float*  sRed = (float*)(smem + 139776);    // 8w x 8h x 16d
    float*  sM   = (float*)(smem + 143872);    // 8w x 8h
    float*  sL   = (float*)(smem + 144128);    // 8w x 8h
    int tid = threadIdx.x, wave = tid >> 5, lane = tid & 31;
    int bi = blockIdx.x, b = bi >> 7, i = bi & 127;
    int colb = lane & 15, khalf = lane >> 4;
    int c = wave, rowl = lane & 15;

    // --- stage augmented weights (WkT||WvT = 128KB contiguous) into LDS ---
#if USE_ASYNC_LDS
    {
        const char* gsrc = (const char*)WkT;   // WvT = WkT + 32768 elems (contiguous)
        for (int off = tid * 16; off < 131072; off += 256 * 16)
            __builtin_amdgcn_global_load_async_to_lds_b128(
                (gv4i_t*)(uintptr_t)(gsrc + off),
                (lv4i_t*)(unsigned)(uintptr_t)(smem + off), 0, 0);
        __builtin_amdgcn_s_wait_asynccnt(0);
    }
#else
    for (int idx = tid; idx < 32768; idx += 256) {
        sWk[idx] = WkT[idx];
        sWv[idx] = WvT[idx];
    }
#endif

    float m[8], l[8], outacc[8];
#pragma unroll
    for (int h = 0; h < 8; ++h) { m[h] = -3.0e38f; l[h] = 0.0f; outacc[h] = 0.0f; }

    for (int jt = 0; jt < 8; ++jt) {
        int j0 = jt * 16;
        __syncthreads();
        // Q tile (16 consecutive rows = contiguous 8KB) + q1 tile
#if USE_ASYNC_LDS
        {
            const char* gq = (const char*)(Qf + (size_t)bi * 16384 + (size_t)j0 * 128);
            for (int off = tid * 16; off < 8192; off += 256 * 16)
                __builtin_amdgcn_global_load_async_to_lds_b128(
                    (gv4i_t*)(uintptr_t)(gq + off),
                    (lv4i_t*)(unsigned)(uintptr_t)((char*)sQ + off), 0, 0);
        }
        for (int idx = tid; idx < 128; idx += 256)
            sQ1[idx] = q1[((size_t)bi * 128 + j0 + (idx >> 3)) * 8 + (idx & 7)];
        __builtin_amdgcn_s_wait_asynccnt(0);
#else
        for (int idx = tid; idx < 2048; idx += 256)
            sQ[idx] = Qf[(size_t)bi * 16384 + (size_t)(j0 + (idx >> 7)) * 128 + (idx & 127)];
        for (int idx = tid; idx < 128; idx += 256)
            sQ1[idx] = q1[((size_t)bi * 128 + j0 + (idx >> 3)) * 8 + (idx & 7)];
#endif
        __syncthreads();

        int j = j0 + rowl;
        const float* a1 = bank1 + ((((size_t)c * 2 + b) * 128 + i) * 128 + j) * 128;
        const float* a0 = bank0 + (((size_t)c * 2 + b) * 128 + j) * 128;
        if (jt < 7) {
            __builtin_prefetch((const void*)(a1 + 2048), 0, 0);
            __builtin_prefetch((const void*)(a0 + 2048), 0, 0);
        }

        v8f accK[8], accV[8];
#pragma unroll
        for (int nt = 0; nt < 8; ++nt) { accK[nt] = vzero8(); accV[nt] = vzero8(); }
        // augmented K dimension = 256: ks 0..3 from bank1 row, ks 4..7 from bank0 row
#pragma unroll
        for (int ks = 0; ks < 8; ++ks) {
            const float* p0 = (ks < 4) ? (a1 + ks * 32 + khalf * 8)
                                       : (a0 + (ks - 4) * 32 + khalf * 8);
            v16bf af;
#pragma unroll
            for (int t = 0; t < 8; ++t) af[t] = (bf16_t)p0[t];
#pragma unroll
            for (int t = 0; t < 8; ++t) af[8 + t] = (bf16_t)p0[16 + t];
#pragma unroll
            for (int nt = 0; nt < 8; ++nt) {
                v16bf bk = load_b_frag(sWk, 256, nt * 16 + colb, khalf, ks * 32);
                accK[nt] = wmma_bf16(af, bk, accK[nt]);
                v16bf bv = load_b_frag(sWv, 256, nt * 16 + colb, khalf, ks * 32);
                accV[nt] = wmma_bf16(af, bv, accV[nt]);
            }
        }
        // per-head scores from accumulators + online softmax + V accumulation
#pragma unroll
        for (int h = 0; h < 8; ++h) {
            float sg[8];
#pragma unroll
            for (int g = 0; g < 8; ++g)
                sg[g] = accK[h][g] * sQ[(khalf * 8 + g) * 128 + h * 16 + colb];
#pragma unroll
            for (int mask = 1; mask < 16; mask <<= 1)
#pragma unroll
                for (int g = 0; g < 8; ++g) sg[g] += __shfl_xor(sg[g], mask, 32);
            float mt = -3.0e38f;
#pragma unroll
            for (int g = 0; g < 8; ++g) {
                sg[g] = (sg[g] + sQ1[(khalf * 8 + g) * 8 + h]) * 0.25f;
                mt = fmaxf(mt, sg[g]);
            }
            mt = fmaxf(mt, __shfl_xor(mt, 16, 32));
            float mnew = fmaxf(m[h], mt);
            float alpha = __expf(m[h] - mnew);
            float esum = 0.0f, oacc = 0.0f;
#pragma unroll
            for (int g = 0; g < 8; ++g) {
                float e = __expf(sg[g] - mnew);
                esum += e;
                oacc += e * accV[h][g];
            }
            esum += __shfl_xor(esum, 16, 32);
            l[h] = l[h] * alpha + esum;
            outacc[h] = outacc[h] * alpha + oacc;
            m[h] = mnew;
        }
    }
    // combine the two row-halves, publish wave partials, merge across waves
#pragma unroll
    for (int h = 0; h < 8; ++h) outacc[h] += __shfl_xor(outacc[h], 16, 32);
    if (lane < 16) {
#pragma unroll
        for (int h = 0; h < 8; ++h) sRed[(wave * 8 + h) * 16 + colb] = outacc[h];
    }
    if (lane == 0) {
#pragma unroll
        for (int h = 0; h < 8; ++h) { sM[wave * 8 + h] = m[h]; sL[wave * 8 + h] = l[h]; }
    }
    __syncthreads();
    if (tid < 128) {
        int h = tid >> 4, d = tid & 15;
        float M = -3.0e38f;
        for (int w = 0; w < 8; ++w) M = fmaxf(M, sM[w * 8 + h]);
        float L = 0.0f, O = 0.0f;
        for (int w = 0; w < 8; ++w) {
            float sc = __expf(sM[w * 8 + h] - M);
            L += sL[w * 8 + h] * sc;
            O += sRed[(w * 8 + h) * 16 + d] * sc;
        }
        attw[(size_t)bi * 128 + tid] = O / L + foldV[b * 128 + tid];
    }
}

// ---------------------------------------------------------------------------
// out = LN(X+Y)*s + o over last dim 128. grid(rows), block(128)
// ---------------------------------------------------------------------------
__global__ void add_ln_kernel(const float* __restrict__ X,
                              const float* __restrict__ Y,
                              const float* __restrict__ s,
                              const float* __restrict__ o,
                              float* __restrict__ out) {
    __shared__ float red[4];
    int r = blockIdx.x, t = threadIdx.x, lane = t & 31, w = t >> 5;
    float v = X[(size_t)r * 128 + t] + Y[(size_t)r * 128 + t];
    float sum = v;
    for (int off = 16; off >= 1; off >>= 1) sum += __shfl_xor(sum, off, 32);
    if (lane == 0) red[w] = sum;
    __syncthreads();
    float mean = (red[0] + red[1] + red[2] + red[3]) * (1.0f / 128.0f);
    float d = v - mean, sq = d * d;
    __syncthreads();
    for (int off = 16; off >= 1; off >>= 1) sq += __shfl_xor(sq, off, 32);
    if (lane == 0) red[w] = sq;
    __syncthreads();
    float var = (red[0] + red[1] + red[2] + red[3]) * (1.0f / 128.0f);
    out[(size_t)r * 128 + t] = d * rsqrtf(var + 1e-5f) * s[t] + o[t];
}

// ---------------------------------------------------------------------------
// edge stage 1: e1 = LN(edge + EL2(relu(EL1(cat)))), cat built on the fly.
// grid(256)=(b,i), block(256), dyn LDS 229376
// ---------------------------------------------------------------------------
__global__ void edge1_kernel(const float* __restrict__ edge,
                             const float* __restrict__ nodeN,
                             const float* __restrict__ graph,
                             const float* EL1w, const float* EL1b,
                             const float* EL2w, const float* EL2b,
                             const float* lns, const float* lno,
                             float* __restrict__ e1out) {
    extern __shared__ char smem[];
    bf16_t* sA  = (bf16_t*)smem;
    bf16_t* sWt = (bf16_t*)(smem + 32768);
    bf16_t* sH  = (bf16_t*)(smem + 98304);
    float*  sO  = (float*)(smem + 163840);
    int tid = threadIdx.x, wave = tid >> 5, lane = tid & 31;
    int bi = blockIdx.x, b = bi >> 7, i = bi & 127;
    int colb = lane & 15, khalf = lane >> 4;

    v8f acc[16];
#pragma unroll
    for (int nt = 0; nt < 16; ++nt) acc[nt] = vzero8();

    for (int kc = 0; kc < 5; ++kc) {
        for (int idx = tid; idx < 16384; idx += 256) {
            int j = idx >> 7, kk = idx & 127;
            float x;
            if (kc == 0)      x = edge[((((size_t)b * 128 + i) * 128) + j) * 128 + kk];
            else if (kc == 1) x = edge[((((size_t)b * 128 + j) * 128) + i) * 128 + kk];
            else if (kc == 2) x = nodeN[((size_t)b * 128 + j) * 128 + kk];
            else if (kc == 3) x = nodeN[((size_t)b * 128 + i) * 128 + kk];
            else              x = graph[(size_t)b * 128 + kk];
            sA[idx] = (bf16_t)x;
        }
        for (int idx = tid; idx < 32768; idx += 256) {
            int c = idx >> 7, kk = idx & 127;
            sWt[idx] = (bf16_t)EL1w[((size_t)(kc * 128 + kk)) * 256 + c];
        }
        __syncthreads();
#pragma unroll
        for (int ks = 0; ks < 4; ++ks) {
            v16bf af = load_a_frag(sA, 128, wave * 16 + (lane & 15), khalf, ks * 32);
#pragma unroll
            for (int nt = 0; nt < 16; ++nt) {
                v16bf bfg = load_b_frag(sWt, 128, nt * 16 + colb, khalf, ks * 32);
                acc[nt] = wmma_bf16(af, bfg, acc[nt]);
            }
        }
        __syncthreads();
    }
#pragma unroll
    for (int nt = 0; nt < 16; ++nt) {
        int col = nt * 16 + colb;
        float bb = EL1b[col];
#pragma unroll
        for (int g = 0; g < 8; ++g) {
            int row = wave * 16 + khalf * 8 + g;
            float vv = acc[nt][g] + bb;
            sH[(size_t)row * 256 + col] = (bf16_t)(vv > 0.0f ? vv : 0.0f);
        }
    }
    __syncthreads();
    for (int idx = tid; idx < 32768; idx += 256) {
        int c = idx >> 8, kk = idx & 255;
        sWt[c * 256 + kk] = (bf16_t)EL2w[(size_t)kk * 128 + c];
    }
    __syncthreads();
    v8f a2[8];
#pragma unroll
    for (int nt = 0; nt < 8; ++nt) a2[nt] = vzero8();
#pragma unroll
    for (int ks = 0; ks < 8; ++ks) {
        v16bf af = load_a_frag(sH, 256, wave * 16 + (lane & 15), khalf, ks * 32);
#pragma unroll
        for (int nt = 0; nt < 8; ++nt) {
            v16bf bfg = load_b_frag(sWt, 256, nt * 16 + colb, khalf, ks * 32);
            a2[nt] = wmma_bf16(af, bfg, a2[nt]);
        }
    }
#pragma unroll
    for (int nt = 0; nt < 8; ++nt) {
        int col = nt * 16 + colb;
        float bb = EL2b[col];
#pragma unroll
        for (int g = 0; g < 8; ++g)
            sO[(wave * 16 + khalf * 8 + g) * 128 + col] = a2[nt][g] + bb;
    }
    __syncthreads();
    if (tid < 128) {
        int j = tid;
        const float* er = &edge[((((size_t)b * 128 + i) * 128) + j) * 128];
        float* out = &e1out[((((size_t)b * 128 + i) * 128) + j) * 128];
        float mean = 0.0f;
        for (int kk = 0; kk < 128; ++kk) mean += sO[j * 128 + kk] + er[kk];
        mean *= (1.0f / 128.0f);
        float var = 0.0f;
        for (int kk = 0; kk < 128; ++kk) {
            float d = sO[j * 128 + kk] + er[kk] - mean;
            var += d * d;
        }
        var *= (1.0f / 128.0f);
        float inv = rsqrtf(var + 1e-5f);
        for (int kk = 0; kk < 128; ++kk) {
            float d = sO[j * 128 + kk] + er[kk] - mean;
            out[kk] = d * inv * lns[kk] + lno[kk];
        }
    }
}

// ---------------------------------------------------------------------------
// edge stage 2: edge_out = LN(e1 + EL4(relu(EL3(e1)))) in place.
// ---------------------------------------------------------------------------
__global__ void edge2_kernel(const float* EL3w, const float* EL3b,
                             const float* EL4w, const float* EL4b,
                             const float* lns, const float* lno,
                             float* __restrict__ e) {
    extern __shared__ char smem[];
    bf16_t* sA  = (bf16_t*)smem;
    bf16_t* sWt = (bf16_t*)(smem + 32768);
    bf16_t* sH  = (bf16_t*)(smem + 98304);
    float*  sO  = (float*)(smem + 163840);
    int tid = threadIdx.x, wave = tid >> 5, lane = tid & 31;
    int bi = blockIdx.x;
    int colb = lane & 15, khalf = lane >> 4;
    float* E = e + (size_t)bi * 16384;

    for (int idx = tid; idx < 16384; idx += 256) sA[idx] = (bf16_t)E[idx];
    for (int idx = tid; idx < 32768; idx += 256) {
        int c = idx >> 7, kk = idx & 127;
        sWt[c * 128 + kk] = (bf16_t)EL3w[(size_t)kk * 256 + c];
    }
    __syncthreads();
    v8f acc[16];
#pragma unroll
    for (int nt = 0; nt < 16; ++nt) acc[nt] = vzero8();
#pragma unroll
    for (int ks = 0; ks < 4; ++ks) {
        v16bf af = load_a_frag(sA, 128, wave * 16 + (lane & 15), khalf, ks * 32);
#pragma unroll
        for (int nt = 0; nt < 16; ++nt) {
            v16bf bfg = load_b_frag(sWt, 128, nt * 16 + colb, khalf, ks * 32);
            acc[nt] = wmma_bf16(af, bfg, acc[nt]);
        }
    }
#pragma unroll
    for (int nt = 0; nt < 16; ++nt) {
        int col = nt * 16 + colb;
        float bb = EL3b[col];
#pragma unroll
        for (int g = 0; g < 8; ++g) {
            int row = wave * 16 + khalf * 8 + g;
            float vv = acc[nt][g] + bb;
            sH[(size_t)row * 256 + col] = (bf16_t)(vv > 0.0f ? vv : 0.0f);
        }
    }
    __syncthreads();
    for (int idx = tid; idx < 32768; idx += 256) {
        int c = idx >> 8, kk = idx & 255;
        sWt[c * 256 + kk] = (bf16_t)EL4w[(size_t)kk * 128 + c];
    }
    __syncthreads();
    v8f a2[8];
#pragma unroll
    for (int nt = 0; nt < 8; ++nt) a2[nt] = vzero8();
#pragma unroll
    for (int ks = 0; ks < 8; ++ks) {
        v16bf af = load_a_frag(sH, 256, wave * 16 + (lane & 15), khalf, ks * 32);
#pragma unroll
        for (int nt = 0; nt < 8; ++nt) {
            v16bf bfg = load_b_frag(sWt, 256, nt * 16 + colb, khalf, ks * 32);
            a2[nt] = wmma_bf16(af, bfg, a2[nt]);
        }
    }
#pragma unroll
    for (int nt = 0; nt < 8; ++nt) {
        int col = nt * 16 + colb;
        float bb = EL4b[col];
#pragma unroll
        for (int g = 0; g < 8; ++g)
            sO[(wave * 16 + khalf * 8 + g) * 128 + col] = a2[nt][g] + bb;
    }
    __syncthreads();
    if (tid < 128) {
        int j = tid;
        float* er = &E[(size_t)j * 128];
        float mean = 0.0f;
        for (int kk = 0; kk < 128; ++kk) mean += sO[j * 128 + kk] + er[kk];
        mean *= (1.0f / 128.0f);
        float var = 0.0f;
        for (int kk = 0; kk < 128; ++kk) {
            float d = sO[j * 128 + kk] + er[kk] - mean;
            var += d * d;
        }
        var *= (1.0f / 128.0f);
        float inv = rsqrtf(var + 1e-5f);
        for (int kk = 0; kk < 128; ++kk) {
            float d = sO[j * 128 + kk] + er[kk] - mean;
            er[kk] = d * inv * lns[kk] + lno[kk];
        }
    }
}

// ---------------------------------------------------------------------------
// Host launcher
// ---------------------------------------------------------------------------
extern "C" void kernel_launch(void* const* d_in, const int* in_sizes, int n_in,
                              void* d_out, int out_size, void* d_ws, size_t ws_size,
                              hipStream_t stream) {
    const float* node  = (const float*)d_in[0];
    const float* edge  = (const float*)d_in[1];
    const float* graph = (const float*)d_in[2];
    const float* bank0 = (const float*)d_in[5];
    const float* bank1 = (const float*)d_in[6];
    const float *Wnq_w=(const float*)d_in[8],  *Wnq_b=(const float*)d_in[9];
    const float *Wnk_w=(const float*)d_in[10], *Wnk_b=(const float*)d_in[11];
    const float *Wnv_w=(const float*)d_in[12], *Wnv_b=(const float*)d_in[13];
    const float *Wgq_w=(const float*)d_in[14], *Wgq_b=(const float*)d_in[15];
    const float *Wgk_w=(const float*)d_in[16], *Wgk_b=(const float*)d_in[17];
    const float *Wgv_w=(const float*)d_in[18], *Wgv_b=(const float*)d_in[19];
    const float *Weq_w=(const float*)d_in[20], *Weq_b=(const float*)d_in[21];
    const float *Wek_w=(const float*)d_in[22], *Wek_b=(const float*)d_in[23];
    const float *Wev_w=(const float*)d_in[24], *Wev_b=(const float*)d_in[25];
    const float *NL1w=(const float*)d_in[26], *NL1b=(const float*)d_in[27];
    const float *NL2w=(const float*)d_in[28], *NL2b=(const float*)d_in[29];
    const float *NL3w=(const float*)d_in[30], *NL3b=(const float*)d_in[31];
    const float *EL1w=(const float*)d_in[32], *EL1b=(const float*)d_in[33];
    const float *EL2w=(const float*)d_in[34], *EL2b=(const float*)d_in[35];
    const float *EL3w=(const float*)d_in[36], *EL3b=(const float*)d_in[37];
    const float *EL4w=(const float*)d_in[38], *EL4b=(const float*)d_in[39];
    const float *NLN1s=(const float*)d_in[40], *NLN1o=(const float*)d_in[41];
    const float *NLN2s=(const float*)d_in[42], *NLN2o=(const float*)d_in[43];
    const float *ELN1s=(const float*)d_in[44], *ELN1o=(const float*)d_in[45];
    const float *ELN2s=(const float*)d_in[46], *ELN2o=(const float*)d_in[47];

    float* ws    = (float*)d_ws;
    float* nQp   = ws;                    // 32768
    float* gQp   = nQp + 32768;           // 256
    float* gKp   = gQp + 256;             // 256
    float* gVp   = gKp + 256;             // 256
    float* Qf    = gVp + 256;             // 4194304
    float* q1    = Qf + 4194304;          // 262144
    bf16_t* WkT  = (bf16_t*)(q1 + 262144);  // 32768 bf16
    bf16_t* WvT  = WkT + 32768;             // 32768 bf16 (contiguous after WkT)
    float* kcolv = (float*)(WvT + 32768); // 256
    float* foldV = kcolv + 256;           // 256
    float* attw  = foldV + 256;           // 32768
    float* tmpH  = attw + 32768;          // 131072
    float* node1 = tmpH + 131072;         // 32768
    float* tmp3  = node1 + 32768;         // 32768

    float* node_out = (float*)d_out;      // 32768
    float* edge_out = node_out + 32768;   // 4194304

    // --- projections / attention prep ---
    graph_proj_kernel<<<2, 128, 0, stream>>>(graph, Wgq_w, Wgq_b, Wgk_w, Wgk_b,
                                             Wgv_w, Wgv_b, gQp, gKp, gVp);
    gemm_wmma_kernel<<<dim3(2, 1), 256, 65536, stream>>>(node, Wnq_w, Wnq_b, nQp,
                                                         256, 128, 128, 0);
    gemm_q_kernel<<<256, 256, 66048, stream>>>(edge, Weq_w, Weq_b, nQp, gQp, Qf);
    prep_attn_kernel<<<32, 256, 0, stream>>>(Wek_w, Wnk_w, Wev_w, Wnv_w,
                                             Wek_b, Wnk_b, Wev_b, Wnv_b,
                                             gKp, gVp, WkT, WvT, kcolv, foldV);
    q1_kernel<<<256, 256, 0, stream>>>(Qf, kcolv, q1);

    // --- fused flash attention (bank1 streamed once) ---
    fused_attn_kernel<<<256, 256, 144384, stream>>>(bank1, bank0, WkT, WvT,
                                                    Qf, q1, foldV, attw);

    // --- node path ---
    gemm_wmma_kernel<<<dim3(2, 1), 256, 65536, stream>>>(attw, NL1w, NL1b, tmpH,
                                                         256, 128, 128, 0);
    add_ln_kernel<<<256, 128, 0, stream>>>(node, tmpH, NLN1s, NLN1o, node1);
    gemm_wmma_kernel<<<dim3(2, 4), 256, 65536, stream>>>(node1, NL2w, NL2b, tmpH,
                                                         256, 512, 128, 1);
    gemm_wmma_kernel<<<dim3(2, 1), 256, 65536, stream>>>(tmpH, NL3w, NL3b, tmp3,
                                                         256, 128, 512, 0);
    add_ln_kernel<<<256, 128, 0, stream>>>(node1, tmp3, NLN2s, NLN2o, node_out);

    // --- edge path ---
    edge1_kernel<<<256, 256, 229376, stream>>>(edge, node_out, graph, EL1w, EL1b,
                                               EL2w, EL2b, ELN1s, ELN1o, edge_out);
    edge2_kernel<<<256, 256, 229376, stream>>>(EL3w, EL3b, EL4w, EL4b,
                                               ELN2s, ELN2o, edge_out);
}
